// VisionTransformer_49606872269178
// MI455X (gfx1250) — compile-verified, matrix operand order checked by reading, not run on hardware
//
#include <hip/hip_runtime.h>
#include <hip/hip_bf16.h>
#include <math.h>

// ---------------------------------------------------------------------------
// CDNA5 / gfx1250 VisionTransformer fused pipeline.
//   out = concat( ssm_q (8*20*3), g (8*5*5*1536), pooled_last0 (8*1536) )
// GEMMs use V_WMMA_F32_16X16X4_F32 (fp32 in / fp32 acc, matches reference
// precision); guarded fallback to confirmed bf16 WMMA.
// ---------------------------------------------------------------------------

typedef __attribute__((ext_vector_type(2))) float v2f;
typedef __attribute__((ext_vector_type(8))) float v8f;

#if defined(__HIP_DEVICE_COMPILE__)
#if __has_builtin(__builtin_amdgcn_wmma_f32_16x16x4_f32)
#define WMMA_PATH 2
#elif __has_builtin(__builtin_amdgcn_wmma_f32_16x16x32_bf16)
#define WMMA_PATH 1
#else
#define WMMA_PATH 0
#endif
#else
#define WMMA_PATH 0
#endif

#if WMMA_PATH == 1
typedef __attribute__((ext_vector_type(16))) __bf16 v16bf;
#endif

#define CDIM 1536
#define BDIM 8
#define TDIM 20
#define NGRP 5
#define KSEQ 5
#define NROWS (BDIM * NGRP * KSEQ)   // 200
#define HWSZ (224 * 224)

// ---------------------------------------------------------------------------
// 1) Fused mean-pool + bbox crop-pool over x : one pass over 96 MB.
//    One block per (b,c,t) plane.  ssm_q[b,t,c] = mean + crop/area.
// ---------------------------------------------------------------------------
__global__ __launch_bounds__(256) void pool_kernel(
    const float* __restrict__ x, const int* __restrict__ bboxes,
    float* __restrict__ ssm_q) {
  const int p = blockIdx.x;          // ((b*3+c)*20+t)
  const int t = p % TDIM;
  const int bc = p / TDIM;
  const int c = bc % 3;
  const int b = bc / 3;
  const int* bb = bboxes + ((b * TDIM + t) * 4);
  const int x1 = bb[0], y1 = bb[1], x2 = bb[2], y2 = bb[3];
  const float* plane = x + (size_t)p * HWSZ;

  float sAll = 0.f, sCrop = 0.f;
  for (int q = threadIdx.x; q < 224 * 56; q += 256) {  // float4 tiles
    const int y = q / 56;
    const int xq = (q % 56) * 4;
    const float4 v = ((const float4*)plane)[q];
    sAll += v.x + v.y + v.z + v.w;
    if (y >= y1 && y < y2) {
      float s = 0.f;
      if (xq + 0 >= x1 && xq + 0 < x2) s += v.x;
      if (xq + 1 >= x1 && xq + 1 < x2) s += v.y;
      if (xq + 2 >= x1 && xq + 2 < x2) s += v.z;
      if (xq + 3 >= x1 && xq + 3 < x2) s += v.w;
      sCrop += s;
    }
  }
  __shared__ float rA[256], rC[256];
  rA[threadIdx.x] = sAll;
  rC[threadIdx.x] = sCrop;
  __syncthreads();
  for (int s = 128; s > 0; s >>= 1) {
    if ((int)threadIdx.x < s) {
      rA[threadIdx.x] += rA[threadIdx.x + s];
      rC[threadIdx.x] += rC[threadIdx.x + s];
    }
    __syncthreads();
  }
  if (threadIdx.x == 0) {
    const float area = (float)((x2 - x1) * (y2 - y1));
    ssm_q[(b * TDIM + t) * 3 + c] = rA[0] * (1.f / (float)HWSZ) + rC[0] / area;
  }
}

// ---------------------------------------------------------------------------
// 2) NT GEMM with WMMA:  C[M,N] = A[M,K] @ W[N,K]^T (+bias[n]) (+resid).
//    One wave per 16x16 output tile.  K multiple of 32.
//    f32 path: V_WMMA_F32_16X16X4_F32, per-lane float2 frags:
//      A frag: lanes 0-15 = rows, half selects K pair  -> A[m][k0+2*half .. +1]
//      B frag (symmetric): W[n][k0+2*half .. +1]
// ---------------------------------------------------------------------------
__global__ __launch_bounds__(256) void gemm_nt_wmma(
    const float* __restrict__ A, const float* __restrict__ W,
    const float* __restrict__ bias, const float* __restrict__ resid,
    float* __restrict__ Cout, int M, int N, int K) {
  const int wave = threadIdx.x >> 5;
  const int lane = threadIdx.x & 31;
  const int tilesN = N >> 4;
  const int tilesM = (M + 15) >> 4;
  const int tile = blockIdx.x * 8 + wave;
  if (tile >= tilesM * tilesN) return;  // wave-uniform exit
  const int tm = tile / tilesN;
  const int tn = tile - tm * tilesN;
  const int half = lane >> 4;
  const int l16 = lane & 15;
  const int mrow = tm * 16 + l16;
  const int ncol = tn * 16 + l16;
  const bool inM = (mrow < M);
  const float* Arow = A + (size_t)(inM ? mrow : 0) * K;
  const float* Wrow = W + (size_t)ncol * K;

  v8f acc = {0.f, 0.f, 0.f, 0.f, 0.f, 0.f, 0.f, 0.f};

#if WMMA_PATH == 2
  __builtin_prefetch(Wrow, 0, 3);  // global_prefetch_b8 of the weight row
  const float* Ap = Arow + 2 * half;
  const float* Wp = Wrow + 2 * half;
#pragma unroll 4
  for (int k0 = 0; k0 < K; k0 += 4) {
    v2f a;
    if (inM) a = *(const v2f*)(Ap + k0);
    else { a[0] = 0.f; a[1] = 0.f; }
    v2f bfrag = *(const v2f*)(Wp + k0);
    acc = __builtin_amdgcn_wmma_f32_16x16x4_f32(
        false, a, false, bfrag, (short)0, acc, false, false);
  }
#elif WMMA_PATH == 1
  for (int k0 = 0; k0 < K; k0 += 32) {
    v16bf a, bfrag;
#pragma unroll
    for (int e = 0; e < 8; ++e) {
      const int kA = k0 + half * 8 + e;
      const float a0 = inM ? Arow[kA] : 0.f;
      const float a1 = inM ? Arow[kA + 16] : 0.f;
      a[e] = (__bf16)a0;
      a[e + 8] = (__bf16)a1;
      bfrag[e] = (__bf16)Wrow[kA];
      bfrag[e + 8] = (__bf16)Wrow[kA + 16];
    }
    acc = __builtin_amdgcn_wmma_f32_16x16x32_bf16(
        false, a, false, bfrag, (short)0, acc, false, false);
  }
#else
  // Host-pass / safety scalar fallback (same result mapping).
  for (int j = 0; j < 8; ++j) {
    const int r = tm * 16 + half * 8 + j;
    if (r < M) {
      float s = 0.f;
      for (int k = 0; k < K; ++k) s += A[(size_t)r * K + k] * Wrow[k];
      acc[j] = s;
    }
  }
#endif

  const float bn = bias ? bias[ncol] : 0.f;
#pragma unroll
  for (int j = 0; j < 8; ++j) {
    const int r = tm * 16 + half * 8 + j;  // C layout: VGPR j -> row j (+8 hi half)
    if (r < M) {
      const size_t off = (size_t)r * N + ncol;
      float v = acc[j] + bn;
      if (resid) v += resid[off];
      Cout[off] = v;
    }
  }
}

// ---------------------------------------------------------------------------
// 3) xt epilogue: += time_embed[t] + timestamp[b]*w_num + b_num
// ---------------------------------------------------------------------------
__global__ __launch_bounds__(256) void xt_finish_kernel(
    float* __restrict__ xt, const float* __restrict__ time_embed,
    const float* __restrict__ timestamp, const float* __restrict__ w_num,
    const float* __restrict__ b_num) {
  const int idx = blockIdx.x * 256 + threadIdx.x;
  if (idx >= BDIM * TDIM * CDIM) return;
  const int c = idx % CDIM;
  const int r = idx / CDIM;
  const int t = r % TDIM;
  const int b = r / TDIM;
  xt[idx] += time_embed[t * CDIM + c] + timestamp[b] * w_num[c] + b_num[c];
}

// pooled_last0[b,c] = (xt[b,18,c] + xt[b,19,c]) / 3   (zero-pad algebra)
__global__ __launch_bounds__(256) void pooled_last_kernel(
    const float* __restrict__ xt, float* __restrict__ outp) {
  const int idx = blockIdx.x * 256 + threadIdx.x;
  if (idx >= BDIM * CDIM) return;
  const int c = idx % CDIM;
  const int b = idx / CDIM;
  outp[idx] = (xt[((size_t)b * TDIM + 18) * CDIM + c] +
               xt[((size_t)b * TDIM + 19) * CDIM + c]) * (1.f / 3.f);
}

// g init: k==0 -> cls_token_swap[n], else xt[b, n*4 + (k-1)]
__global__ __launch_bounds__(256) void ginit_kernel(
    const float* __restrict__ xt, const float* __restrict__ cls,
    float* __restrict__ g) {
  const int idx = blockIdx.x * 256 + threadIdx.x;
  if (idx >= NROWS * CDIM) return;
  const int c = idx % CDIM;
  const int row = idx / CDIM;
  const int k = row % KSEQ;
  const int n = (row / KSEQ) % NGRP;
  const int b = row / (KSEQ * NGRP);
  g[idx] = (k == 0) ? cls[n * CDIM + c]
                    : xt[((size_t)(b * TDIM + n * 4 + (k - 1))) * CDIM + c];
}

// ---------------------------------------------------------------------------
// 4) LayerNorm over 1536-wide rows (one block per row).
// ---------------------------------------------------------------------------
__global__ __launch_bounds__(256) void ln_kernel(
    const float* __restrict__ g, const float* __restrict__ w,
    const float* __restrict__ bvec, float* __restrict__ h) {
  const int r = blockIdx.x;
  const float* row = g + (size_t)r * CDIM;
  float s = 0.f, ss = 0.f;
  for (int c = threadIdx.x; c < CDIM; c += 256) {
    const float v = row[c];
    s += v;
    ss += v * v;
  }
  __shared__ float rs[256], rss[256];
  rs[threadIdx.x] = s;
  rss[threadIdx.x] = ss;
  __syncthreads();
  for (int st = 128; st > 0; st >>= 1) {
    if ((int)threadIdx.x < st) {
      rs[threadIdx.x] += rs[threadIdx.x + st];
      rss[threadIdx.x] += rss[threadIdx.x + st];
    }
    __syncthreads();
  }
  const float mean = rs[0] * (1.f / (float)CDIM);
  const float var = rss[0] * (1.f / (float)CDIM) - mean * mean;
  const float inv = rsqrtf(var + 1e-5f);
  for (int c = threadIdx.x; c < CDIM; c += 256)
    h[(size_t)r * CDIM + c] = (row[c] - mean) * inv * w[c] + bvec[c];
}

// ---------------------------------------------------------------------------
// 5) Attention, k=5, hd=128: one thread per (seq, head, query-row).
//    qkv row layout: col = (comp*12 + head)*128 + d.
// ---------------------------------------------------------------------------
__global__ __launch_bounds__(128) void attn_kernel(
    const float* __restrict__ qkv, float* __restrict__ o) {
  const int tid = blockIdx.x * 128 + threadIdx.x;
  if (tid >= 40 * 12 * 5) return;
  const int i = tid % 5;
  const int h = (tid / 5) % 12;
  const int s = tid / 60;
  const float scale = 0.08838834764831845f;  // 128^-0.5

  const float* qi = qkv + (size_t)(s * 5 + i) * (3 * CDIM) + (0 * 12 + h) * 128;
  float sc[5];
  float mx = -1e30f;
  for (int j = 0; j < 5; ++j) {
    const float* kj = qkv + (size_t)(s * 5 + j) * (3 * CDIM) + (12 + h) * 128;
    float d = 0.f;
    for (int q4 = 0; q4 < 32; ++q4) {
      const float4 a = ((const float4*)qi)[q4];
      const float4 bk = ((const float4*)kj)[q4];
      d += a.x * bk.x + a.y * bk.y + a.z * bk.z + a.w * bk.w;
    }
    sc[j] = d * scale;
    mx = fmaxf(mx, sc[j]);
  }
  float den = 0.f;
  for (int j = 0; j < 5; ++j) {
    sc[j] = __expf(sc[j] - mx);
    den += sc[j];
  }
  const float rden = 1.f / den;
  float* orow = o + (size_t)(s * 5 + i) * CDIM + h * 128;
  for (int q4 = 0; q4 < 32; ++q4) {
    float4 av = {0.f, 0.f, 0.f, 0.f};
    for (int j = 0; j < 5; ++j) {
      const float4 vv =
          ((const float4*)(qkv + (size_t)(s * 5 + j) * (3 * CDIM) +
                           (24 + h) * 128))[q4];
      const float pj = sc[j] * rden;
      av.x += pj * vv.x; av.y += pj * vv.y;
      av.z += pj * vv.z; av.w += pj * vv.w;
    }
    ((float4*)orow)[q4] = av;
  }
}

// commit + roll: g[b,n,0,:] = gnext[b,(n-1)%5,0,:]; else copy.
__global__ __launch_bounds__(256) void commit_roll_kernel(
    const float* __restrict__ gnext, float* __restrict__ g) {
  const int idx = blockIdx.x * 256 + threadIdx.x;
  if (idx >= NROWS * CDIM) return;
  const int c = idx % CDIM;
  const int row = idx / CDIM;
  const int k = row % KSEQ;
  const int n = (row / KSEQ) % NGRP;
  const int b = row / (KSEQ * NGRP);
  if (k == 0) {
    const int src = (b * NGRP + ((n + NGRP - 1) % NGRP)) * KSEQ;
    g[idx] = gnext[(size_t)src * CDIM + c];
  } else {
    g[idx] = gnext[idx];
  }
}

// ---------------------------------------------------------------------------
// Launch
// ---------------------------------------------------------------------------
static inline int gemm_blocks(int M, int N) {
  const int tiles = ((M + 15) / 16) * (N / 16);
  return (tiles + 7) / 8;  // 8 waves (16x16 tiles) per 256-thread block
}

extern "C" void kernel_launch(void* const* d_in, const int* in_sizes, int n_in,
                              void* d_out, int out_size, void* d_ws,
                              size_t ws_size, hipStream_t stream) {
  const float* x = (const float*)d_in[0];
  const float* timestamp = (const float*)d_in[1];
  const int* bboxes = (const int*)d_in[2];
  const float* feats = (const float*)d_in[3];
  const float* w_reduce = (const float*)d_in[4];
  const float* b_reduce = (const float*)d_in[5];
  const float* w_num = (const float*)d_in[6];
  const float* b_num = (const float*)d_in[7];
  const float* time_embed = (const float*)d_in[8];
  const float* cls = (const float*)d_in[9];
  const float* ln_w = (const float*)d_in[10];
  const float* ln_b = (const float*)d_in[11];
  const float* w_qkv = (const float*)d_in[12];
  const float* w_proj = (const float*)d_in[13];
  const float* b_proj = (const float*)d_in[14];
  const float* w_tfc = (const float*)d_in[15];
  const float* b_tfc = (const float*)d_in[16];

  float* out = (float*)d_out;
  float* ssm_q = out;                              // 480
  float* g = out + 480;                            // 307200
  float* pooled = out + 480 + NROWS * CDIM;        // 12288

  float* ws = (float*)d_ws;
  float* xt = ws;                                  // 160*1536
  float* h = xt + BDIM * TDIM * CDIM;              // 200*1536
  float* qkv = h + NROWS * CDIM;                   // 200*4608
  float* o = qkv + (size_t)NROWS * 3 * CDIM;       // 200*1536
  float* p = o + NROWS * CDIM;                     // 200*1536
  float* gnext = p + NROWS * CDIM;                 // 200*1536

  // 1) ssm_q (bandwidth-dominant pass over x)
  pool_kernel<<<BDIM * 3 * TDIM, 256, 0, stream>>>(x, bboxes, ssm_q);

  // 2) xt = feats @ w_reduce^T + b_reduce (+ embeds)
  gemm_nt_wmma<<<gemm_blocks(BDIM * TDIM, CDIM), 256, 0, stream>>>(
      feats, w_reduce, b_reduce, nullptr, xt, BDIM * TDIM, CDIM, CDIM);
  xt_finish_kernel<<<(BDIM * TDIM * CDIM + 255) / 256, 256, 0, stream>>>(
      xt, time_embed, timestamp, w_num, b_num);
  pooled_last_kernel<<<(BDIM * CDIM + 255) / 256, 256, 0, stream>>>(xt, pooled);

  // 3) init g, then 4 transformer iterations
  ginit_kernel<<<(NROWS * CDIM + 255) / 256, 256, 0, stream>>>(xt, cls, g);

  for (int it = 0; it < 4; ++it) {
    ln_kernel<<<NROWS, 256, 0, stream>>>(g, ln_w, ln_b, h);
    gemm_nt_wmma<<<gemm_blocks(NROWS, 3 * CDIM), 256, 0, stream>>>(
        h, w_qkv, nullptr, nullptr, qkv, NROWS, 3 * CDIM, CDIM);
    attn_kernel<<<(40 * 12 * 5 + 127) / 128, 128, 0, stream>>>(qkv, o);
    gemm_nt_wmma<<<gemm_blocks(NROWS, CDIM), 256, 0, stream>>>(
        o, w_proj, b_proj, nullptr, p, NROWS, CDIM, CDIM);
    gemm_nt_wmma<<<gemm_blocks(NROWS, CDIM), 256, 0, stream>>>(
        p, w_tfc, b_tfc, g, gnext, NROWS, CDIM, CDIM);
    commit_roll_kernel<<<(NROWS * CDIM + 255) / 256, 256, 0, stream>>>(gnext, g);
  }
}
